// GCL_87617332838483
// MI455X (gfx1250) — matrix operand order repaired and checked by reading, not compiled, not run on hardware
//
#include <hip/hip_runtime.h>
#include <hip/hip_bf16.h>

typedef __bf16 bf16_t;
typedef __attribute__((ext_vector_type(16))) __bf16 v16bf;
typedef __attribute__((ext_vector_type(8)))  float  v8f;
typedef __attribute__((address_space(3))) unsigned short lds_us_t;

union AFrag { v16bf v; unsigned int u[8]; };
union CFrag { v8f  v; float f[8]; };

#define LDS_STRIDE 40   // shorts per LDS row: 32 data + 8 pad -> 80B rows (16B aligned)

// Async direct-to-LDS 16B copy: per-lane LDS byte offset + per-lane global address.
#define ASYNC_B128(ldsoff, gptr)                                     \
  asm volatile("global_load_async_to_lds_b128 %0, %1, off"           \
               :: "v"(ldsoff), "v"(gptr) : "memory")

// ---- segment providers -----------------------------------------------------
// LSTM: 2 segments (input-hidden + hidden-hidden), resolved by cselect.
struct SegLstm {
  const bf16_t* A0; const bf16_t* B0;
  const bf16_t* A1; const bf16_t* B1;
  __device__ __forceinline__ void get(int s, const bf16_t*& a, const bf16_t*& b) const {
    a = (s == 0) ? A0 : A1;
    b = (s == 0) ? B0 : B1;
  }
};

// Conv: segments = valid 5x5 taps for output position t = blockIdx.z,
// derived arithmetically (kh in [max(0,2-h), min(4,5-h)], kw in {2-w, 3-w}).
struct SegConv {
  const bf16_t* xseq;   // [tin][b*1024+ci] bf16
  const bf16_t* cwt;    // [tap][co][ci]    bf16
  __device__ __forceinline__ void get(int s, const bf16_t*& a, const bf16_t*& b) const {
    const int t = blockIdx.z;
    const int h = t >> 1, w = t & 1;
    const int khmin = (2 - h) > 0 ? (2 - h) : 0;
    const int kh = khmin + (s >> 1);
    const int kw = (2 - w) + (s & 1);
    const int tin = (h + kh - 2) * 2 + (w + kw - 2);
    const int tap = kh * 5 + kw;
    a = xseq + (size_t)tin * 1048576;
    b = cwt  + (size_t)tap * 1048576;
  }
};

// ---- shared WMMA GEMM body: C[1024][ldc] = sum_s A_s[1024][1024] * B_s[ldc][1024]^T
// A row-major [m][k], B n-major [n][k], both row stride 1024 bf16.
// Single compute site; K-loop unrolling disabled so the allocator keeps each
// accumulator tile pinned (D tied to C) instead of emitting v_mov/v_nop storms.
template <typename SEG>
__device__ __forceinline__ void gemm_body(const SEG& segs, int nseg, float* C, int ldc) {
  __shared__ __align__(16) unsigned short As[2][128 * LDS_STRIDE];
  __shared__ __align__(16) unsigned short Bs[2][128 * LDS_STRIDE];

  const int tid  = threadIdx.x;
  const int lane = tid & 31;
  const int wave = tid >> 5;        // 0..7
  const int mw   = wave >> 1;       // 0..3 -> M offset mw*32
  const int nw   = wave & 1;        // 0..1 -> N offset nw*64
  const int m0   = blockIdx.y * 128;
  const int n0   = blockIdx.x * 128;
  const int half = lane >> 4;       // 0: lanes 0-15, 1: lanes 16-31
  const int l16  = lane & 15;

  // staging: each thread copies 16 contiguous bf16 of one row (2 x b128)
  const int srow = tid >> 1;            // 0..127
  const int skh  = (tid & 1) << 4;      // 0 or 16
  const size_t rowOffA = (size_t)(m0 + srow) * 1024 + skh;
  const size_t rowOffB = (size_t)(n0 + srow) * 1024 + skh;

  unsigned ldsA[2], ldsB[2];
  ldsA[0] = (unsigned)(unsigned long long)(lds_us_t*)&As[0][srow * LDS_STRIDE + skh];
  ldsA[1] = (unsigned)(unsigned long long)(lds_us_t*)&As[1][srow * LDS_STRIDE + skh];
  ldsB[0] = (unsigned)(unsigned long long)(lds_us_t*)&Bs[0][srow * LDS_STRIDE + skh];
  ldsB[1] = (unsigned)(unsigned long long)(lds_us_t*)&Bs[1][srow * LDS_STRIDE + skh];

  CFrag acc[2][4];
#pragma unroll
  for (int i = 0; i < 2; ++i)
#pragma unroll
    for (int j = 0; j < 4; ++j)
#pragma unroll
      for (int r = 0; r < 8; ++r) acc[i][j].f[r] = 0.0f;

  auto issue = [&](const bf16_t* a, const bf16_t* b, int buf) {
    ASYNC_B128(ldsA[buf], a);
    ASYNC_B128(ldsA[buf] + 16u, a + 8);
    ASYNC_B128(ldsB[buf], b);
    ASYNC_B128(ldsB[buf] + 16u, b + 8);
  };

  const int total = nseg * 32;   // K-steps of 32 across all segments

  // prologue: async stage of tile 0 into buffer 0
  {
    const bf16_t *a, *b;
    segs.get(0, a, b);
    issue(a + rowOffA, b + rowOffB, 0);
  }

  int cur = 0;
#pragma clang loop unroll(disable)
  for (int it = 0; it < total; ++it) {
    // wait for this wave's async copies of buffer `cur`, then sync the workgroup
    asm volatile("s_wait_asynccnt 0x0" ::: "memory");
    __syncthreads();

    // kick off async stage of the next tile into the other buffer (overlaps compute)
    if (it + 1 < total) {
      const int nit  = it + 1;
      const int koff = (nit & 31) * 32;
      const bf16_t *an, *bn;
      segs.get(nit >> 5, an, bn);
      issue(an + rowOffA + koff, bn + rowOffB + koff, cur ^ 1);
    }

    // ---- load fragments per ISA 16-bit A/B layouts (wave32), single compute site
    AFrag af[2], bfx[4];
#pragma unroll
    for (int i = 0; i < 2; ++i) {
      const int arow = mw * 32 + i * 16 + l16;
      const unsigned short* ap = &As[cur][arow * LDS_STRIDE];
#pragma unroll
      for (int v = 0; v < 8; ++v) {
        const int kp = (v < 4 ? 2 * v : 16 + 2 * (v - 4)) + half * 8;
        af[i].u[v] = *(const unsigned int*)&ap[kp];
      }
    }
#pragma unroll
    for (int j = 0; j < 4; ++j) {
      const int col = nw * 64 + j * 16 + l16;
      const unsigned short* bp = &Bs[cur][col * LDS_STRIDE + half * 8];
#pragma unroll
      for (int v = 0; v < 8; ++v)
        bfx[j].u[v] = *(const unsigned int*)&bp[v < 4 ? 2 * v : 16 + 2 * (v - 4)];
    }

#pragma unroll
    for (int i = 0; i < 2; ++i)
#pragma unroll
      for (int j = 0; j < 4; ++j)
        acc[i][j].v = __builtin_amdgcn_wmma_f32_16x16x32_bf16(
            false, af[i].v, false, bfx[j].v, (short)0, acc[i][j].v, false, false);

    cur ^= 1;
  }

  // ---- epilogue: C/D layout -> VGPR r holds (M = r + 8*half, N = l16)
#pragma unroll
  for (int i = 0; i < 2; ++i) {
    const int rbase = m0 + mw * 32 + i * 16 + half * 8;
#pragma unroll
    for (int j = 0; j < 4; ++j) {
      const int col = n0 + nw * 64 + j * 16 + l16;
      float* cp = C + (size_t)rbase * ldc + col;
#pragma unroll
      for (int r = 0; r < 8; ++r)
        cp[(size_t)r * ldc] = acc[i][j].f[r];
    }
  }
}

__launch_bounds__(256)
__global__ void gemm_lstm(SegLstm segs, float* C) {
  gemm_body(segs, 2, C, 4096);
}

__launch_bounds__(256)
__global__ void gemm_conv(SegConv segs, float* yconv) {
  const int t = blockIdx.z;
  const int h = t >> 1;
  const int khmin = (2 - h) > 0 ? (2 - h) : 0;
  const int khmax = (5 - h) < 4 ? (5 - h) : 4;
  gemm_body(segs, (khmax - khmin + 1) * 2, yconv + (size_t)t * 1048576, 1024);
}

// ---- x [B][C][8] f32 -> xseq [t][b*1024+c] bf16
__global__ void pack_x(const float* __restrict__ x, bf16_t* __restrict__ xseq) {
  const int idx = blockIdx.x * 256 + threadIdx.x;     // b*1024+c, 1M threads
  const float4* p = (const float4*)(x + (size_t)idx * 8);
  float4 a = p[0], b = p[1];
  float v[8] = {a.x, a.y, a.z, a.w, b.x, b.y, b.z, b.w};
#pragma unroll
  for (int t = 0; t < 8; ++t)
    xseq[(size_t)t * 1048576 + idx] = (bf16_t)v[t];
}

// ---- conv_w [co][ci][5][5] f32 -> cwt [tap][co][ci] bf16
__global__ void pack_convw(const float* __restrict__ w, bf16_t* __restrict__ out) {
  const int idx = blockIdx.x * 256 + threadIdx.x;     // 26,214,400 threads
  const int ci  = idx & 1023;
  const int co  = (idx >> 10) & 1023;
  const int tap = idx >> 20;                          // 0..24
  out[idx] = (bf16_t)w[(size_t)(co * 1024 + ci) * 25 + tap];
}

__global__ void cast_f32_to_bf16(const float* __restrict__ in, bf16_t* __restrict__ out, int n) {
  const int i = blockIdx.x * 256 + threadIdx.x;
  if (i < n) out[i] = (bf16_t)in[i];
}

// ---- deterministic two-stage BN stats over rows of yconv [8192][1024]
__global__ void bn_stats_partial(const float* __restrict__ y,
                                 float* __restrict__ psum, float* __restrict__ psumsq) {
  const int c  = blockIdx.x * 256 + threadIdx.x;   // gridDim.x = 4
  const int rg = blockIdx.y;                       // 32 row groups of 256
  const float* p = y + (size_t)rg * 256 * 1024 + c;
  float s = 0.f, ss = 0.f;
  for (int r = 0; r < 256; ++r) { float v = p[(size_t)r * 1024]; s += v; ss += v * v; }
  psum[rg * 1024 + c] = s;
  psumsq[rg * 1024 + c] = ss;
}

__global__ void bn_finalize(const float* __restrict__ psum, const float* __restrict__ psumsq,
                            float* __restrict__ mean, float* __restrict__ invstd) {
  const int c = blockIdx.x * 256 + threadIdx.x;
  float s = 0.f, ss = 0.f;
  for (int r = 0; r < 32; ++r) { s += psum[r * 1024 + c]; ss += psumsq[r * 1024 + c]; }
  const float m = s * (1.0f / 8192.0f);
  const float v = ss * (1.0f / 8192.0f) - m * m;
  mean[c]   = m;
  invstd[c] = rsqrtf(v + 1e-5f);
}

// ---- normalize + ReLU + pack to bf16 sequence [t][b][c]
__global__ void bn_apply_relu_pack(const float* __restrict__ y,
                                   const float* __restrict__ mean, const float* __restrict__ invstd,
                                   const float* __restrict__ gamma, const float* __restrict__ beta,
                                   bf16_t* __restrict__ seq) {
  const int idx = blockIdx.x * 256 + threadIdx.x;  // 8,388,608 threads
  const int c = idx & 1023;
  const float v = (y[idx] - mean[c]) * invstd[c] * gamma[c] + beta[c];
  seq[idx] = (bf16_t)fmaxf(v, 0.0f);
}

// ---- LSTM cell elementwise: gates [b][4096] f32 (i,f,g,o chunks)
__global__ void lstm_cell(const float* __restrict__ gates,
                          const float* __restrict__ bih, const float* __restrict__ bhh,
                          float* __restrict__ cst, bf16_t* __restrict__ hbf,
                          float* __restrict__ out, int t) {
  const int idx = blockIdx.x * 256 + threadIdx.x;  // 1M threads
  const int b = idx >> 10, c = idx & 1023;
  const float* g = gates + (size_t)b * 4096;
  const float gi = g[c]        + bih[c]        + bhh[c];
  const float gf = g[1024 + c] + bih[1024 + c] + bhh[1024 + c];
  const float gc = g[2048 + c] + bih[2048 + c] + bhh[2048 + c];
  const float go = g[3072 + c] + bih[3072 + c] + bhh[3072 + c];
  const float i = 1.0f / (1.0f + expf(-gi));
  const float f = 1.0f / (1.0f + expf(-gf));
  const float o = 1.0f / (1.0f + expf(-go));
  const float gg = tanhf(gc);
  const float cn = f * cst[idx] + i * gg;
  const float hn = o * tanhf(cn);
  cst[idx] = cn;
  hbf[idx] = (bf16_t)hn;
  if (out) out[(size_t)idx * 8 + t] = hn;   // out[b][c][t]
}

extern "C" void kernel_launch(void* const* d_in, const int* in_sizes, int n_in,
                              void* d_out, int out_size, void* d_ws, size_t ws_size,
                              hipStream_t stream) {
  (void)in_sizes; (void)n_in; (void)out_size; (void)ws_size;
  const float* x      = (const float*)d_in[0];
  const float* conv_w = (const float*)d_in[1];
  const float* gamma  = (const float*)d_in[3];
  const float* beta   = (const float*)d_in[4];
  const float* w_ih0  = (const float*)d_in[5];
  const float* w_hh0  = (const float*)d_in[6];
  const float* b_ih0  = (const float*)d_in[7];
  const float* b_hh0  = (const float*)d_in[8];
  const float* w_ih1  = (const float*)d_in[9];
  const float* w_hh1  = (const float*)d_in[10];
  const float* b_ih1  = (const float*)d_in[11];
  const float* b_hh1  = (const float*)d_in[12];
  float* out = (float*)d_out;

  // ---- carve workspace (all sizes 256-byte multiples)
  char* ws = (char*)d_ws;
  auto carve = [&](size_t bytes) { char* p = ws; ws += (bytes + 255) & ~(size_t)255; return p; };
  bf16_t* xseq   = (bf16_t*)carve(8ull  * 1048576 * 2);  // [t][b*1024+c]
  bf16_t* cwt    = (bf16_t*)carve(25ull * 1048576 * 2);  // [tap][co][ci]
  bf16_t* wih0b  = (bf16_t*)carve(4ull  * 1048576 * 2);  // [4096][1024] bf16
  bf16_t* whh0b  = (bf16_t*)carve(4ull  * 1048576 * 2);
  bf16_t* wih1b  = (bf16_t*)carve(4ull  * 1048576 * 2);
  bf16_t* whh1b  = (bf16_t*)carve(4ull  * 1048576 * 2);
  float*  yconv  = (float*) carve(8ull  * 1048576 * 4);  // [t][b][co]
  bf16_t* seqbf  = (bf16_t*)carve(8ull  * 1048576 * 2);  // post-BN sequence
  float*  gates  = (float*) carve(1024ull * 4096 * 4);
  float*  c0     = (float*) carve(1048576ull * 4);
  float*  c1     = (float*) carve(1048576ull * 4);
  bf16_t* h0bf   = (bf16_t*)carve(1048576ull * 2);
  bf16_t* h1bf   = (bf16_t*)carve(1048576ull * 2);
  float*  psum   = (float*) carve(32ull * 1024 * 4);
  float*  psumsq = (float*) carve(32ull * 1024 * 4);
  float*  meanv  = (float*) carve(1024 * 4);
  float*  invstd = (float*) carve(1024 * 4);

  // ---- packing / casting
  pack_x    <<<1048576 / 256, 256, 0, stream>>>(x, xseq);
  pack_convw<<<26214400 / 256, 256, 0, stream>>>(conv_w, cwt);
  cast_f32_to_bf16<<<4194304 / 256, 256, 0, stream>>>(w_ih0, wih0b, 4194304);
  cast_f32_to_bf16<<<4194304 / 256, 256, 0, stream>>>(w_hh0, whh0b, 4194304);
  cast_f32_to_bf16<<<4194304 / 256, 256, 0, stream>>>(w_ih1, wih1b, 4194304);
  cast_f32_to_bf16<<<4194304 / 256, 256, 0, stream>>>(w_hh1, whh1b, 4194304);

  // zero LSTM state (c0,c1,h0bf,h1bf are contiguous in the carve order)
  hipMemsetAsync(c0, 0, 1048576ull * 4 * 2 + 1048576ull * 2 * 2, stream);

  // ---- conv: all 8 output positions in one launch, taps derived in-kernel
  {
    SegConv sc{xseq, cwt};
    gemm_conv<<<dim3(8, 8, 8), 256, 0, stream>>>(sc, yconv);
  }

  // ---- batchnorm (train-mode batch stats) + ReLU + bf16 pack
  bn_stats_partial<<<dim3(4, 32), 256, 0, stream>>>(yconv, psum, psumsq);
  bn_finalize<<<4, 256, 0, stream>>>(psum, psumsq, meanv, invstd);
  bn_apply_relu_pack<<<8388608 / 256, 256, 0, stream>>>(yconv, meanv, invstd, gamma, beta, seqbf);

  // ---- 2-layer LSTM over 8 timesteps: fused (ih + hh) GEMM per layer-step
  for (int t = 0; t < 8; ++t) {
    SegLstm g0{seqbf + (size_t)t * 1048576, wih0b, h0bf, whh0b};
    gemm_lstm<<<dim3(32, 8), 256, 0, stream>>>(g0, gates);
    lstm_cell<<<1048576 / 256, 256, 0, stream>>>(gates, b_ih0, b_hh0, c0, h0bf, nullptr, t);

    SegLstm g1{h0bf, wih1b, h1bf, whh1b};
    gemm_lstm<<<dim3(32, 8), 256, 0, stream>>>(g1, gates);
    lstm_cell<<<1048576 / 256, 256, 0, stream>>>(gates, b_ih1, b_hh1, c1, h1bf, out, t);
  }
}
